// MoELayer_17059610100270
// MI455X (gfx1250) — matrix operand order repaired and checked by reading, not compiled, not run on hardware
//
#include <hip/hip_runtime.h>
#include <hip/hip_bf16.h>
#include <math.h>

#define D_MODEL 1024
#define D_FF    4096
#define NE      8
#define T_TOKENS 4096      // B*S = 2*2048
#define NSLOTS  (T_TOKENS*2)
#define AST     40         // LDS row stride in bf16 elems (80B, 16B-aligned rows)

typedef __attribute__((ext_vector_type(16))) __bf16 v16bf;
typedef __attribute__((ext_vector_type(8)))  float  v8f;

union FragB16 { uint4 u[2]; v16bf v; };

__device__ __forceinline__ unsigned short f32_to_bf16(float f) {
  unsigned u = __float_as_uint(f);
  unsigned r = u + 0x7FFFu + ((u >> 16) & 1u);   // round-to-nearest-even
  return (unsigned short)(r >> 16);
}

// CDNA5 packed f32->bf16 convert (builtin not exposed by this clang; the
// instruction family exists on gfx1250, so use inline asm).
__device__ __forceinline__ unsigned pack2(float a, float b) {
  unsigned r;
  asm("v_cvt_pk_bf16_f32 %0, %1, %2" : "=v"(r) : "v"(a), "v"(b));
  return r;
}

#if defined(__has_builtin)
#  if __has_builtin(__builtin_amdgcn_global_load_async_to_lds_b128) && \
      __has_builtin(__builtin_amdgcn_s_wait_asynccnt)
#    define USE_ASYNC_LDS 1
#  endif
#endif
#ifndef USE_ASYNC_LDS
#  define USE_ASYNC_LDS 0
#endif

#if USE_ASYNC_LDS
typedef int v4i_gcc __attribute__((vector_size(16)));
typedef __attribute__((address_space(1))) v4i_gcc gv4i;  // global int4*
typedef __attribute__((address_space(3))) v4i_gcc lv4i;  // LDS int4*
#endif

// 32-byte staging copy: global bf16 -> LDS (async DMA when available)
__device__ __forceinline__ void stage32B(const unsigned short* src,
                                         unsigned short* lds) {
#if USE_ASYNC_LDS
  lv4i* d = (lv4i*)lds;
  __builtin_amdgcn_global_load_async_to_lds_b128((gv4i*)src, d, 0,  0);
  __builtin_amdgcn_global_load_async_to_lds_b128((gv4i*)src, d, 16, 0);
#else
  uint4 q0 = *(const uint4*)src;
  uint4 q1 = *(const uint4*)(src + 8);
  uint4* d = (uint4*)lds; d[0] = q0; d[1] = q1;
#endif
}

__device__ __forceinline__ void stage_wait() {
#if USE_ASYNC_LDS
  __builtin_amdgcn_s_wait_asynccnt(0);
#endif
}

// ---------------------------------------------------------------- router ----
__global__ void __launch_bounds__(256)
router_kernel(const float* __restrict__ x,
              const float* __restrict__ gate_w,
              const float* __restrict__ gate_b,
              int*   __restrict__ counts,
              int*   __restrict__ tokenList,   // [NE][T] slot ids
              float* __restrict__ wSlot,       // [NSLOTS]
              float* __restrict__ partialUsage)// [numBlocks][NE]
{
  __shared__ float probsLds[8][NE];
  int lane = threadIdx.x & 31;
  int wave = threadIdx.x >> 5;
  int token = blockIdx.x * 8 + wave;

  const float* xp = x + (size_t)token * D_MODEL;
  float xr[32];
#pragma unroll
  for (int i = 0; i < 32; ++i) xr[i] = xp[lane + 32 * i];

  float logit[NE];
#pragma unroll
  for (int e = 0; e < NE; ++e) {
    const float* gw = gate_w + e * D_MODEL;
    float acc = 0.f;
#pragma unroll
    for (int i = 0; i < 32; ++i) acc = fmaf(xr[i], gw[lane + 32 * i], acc);
#pragma unroll
    for (int off = 16; off > 0; off >>= 1) acc += __shfl_xor(acc, off, 32);
    logit[e] = acc + gate_b[e];
  }
  // top-2
  int i1 = 0; float m1 = logit[0];
#pragma unroll
  for (int e = 1; e < NE; ++e) if (logit[e] > m1) { m1 = logit[e]; i1 = e; }
  int i2 = (i1 == 0) ? 1 : 0; float m2 = logit[i2];
#pragma unroll
  for (int e = 0; e < NE; ++e) if (e != i1 && logit[e] > m2) { m2 = logit[e]; i2 = e; }
  float e2 = __expf(m2 - m1);
  float wTop = 1.f / (1.f + e2);
  float wSec = e2 * wTop;
  // full softmax for usage
  float p[NE], denom = 0.f;
#pragma unroll
  for (int e = 0; e < NE; ++e) { p[e] = __expf(logit[e] - m1); denom += p[e]; }
  float inv = 1.f / denom;
  if (lane < NE) probsLds[wave][lane] = p[lane] * inv;
  if (lane == 0) {
    int pos1 = atomicAdd(&counts[i1], 1);
    tokenList[i1 * T_TOKENS + pos1] = token * 2;
    wSlot[token * 2] = wTop;
    int pos2 = atomicAdd(&counts[i2], 1);
    tokenList[i2 * T_TOKENS + pos2] = token * 2 + 1;
    wSlot[token * 2 + 1] = wSec;
  }
  __syncthreads();
  if (threadIdx.x < NE) {
    float s = 0.f;
#pragma unroll
    for (int wv = 0; wv < 8; ++wv) s += probsLds[wv][threadIdx.x];
    partialUsage[blockIdx.x * NE + threadIdx.x] = s;
  }
}

__global__ void loss_kernel(const float* __restrict__ partialUsage,
                            float* __restrict__ outLoss, int nblocks)
{
  int lane = threadIdx.x;
  float s = 0.f;
  if (lane < NE) {
    for (int b = 0; b < nblocks; ++b) s += partialUsage[b * NE + lane];
    s *= (1.0f / (float)T_TOKENS);
  }
  float sq = (lane < NE) ? s * s : 0.f;
#pragma unroll
  for (int off = 4; off > 0; off >>= 1) sq += __shfl_xor(sq, off, 32);
  if (lane == 0) *outLoss = (float)NE * sq - 1.0f;
}

// ----------------------------------------------------- prep: x fp32 -> bf16 -
__global__ void __launch_bounds__(256)
convert_x_kernel(const float* __restrict__ in, unsigned short* __restrict__ outv)
{
  size_t i = ((size_t)blockIdx.x * 256 + threadIdx.x) * 4;  // 4 floats each
  float4 v = *(const float4*)(in + i);
  uint2 r; r.x = pack2(v.x, v.y); r.y = pack2(v.z, v.w);
  *(uint2*)(outv + i) = r;
}

// ------------------------------- prep: per-expert transpose + fp32 -> bf16 --
// in:  [E][R][C] fp32     outT: [E][C][R] bf16     (R, C multiples of 32)
__global__ void __launch_bounds__(256)
transpose_convert_kernel(const float* __restrict__ in,
                         unsigned short* __restrict__ outT,
                         int R, int C)
{
  __shared__ unsigned short tile[32][33];
  int e = blockIdx.z;
  const float*    src = in  + (size_t)e * R * C;
  unsigned short* dst = outT + (size_t)e * R * C;
  int tx = threadIdx.x & 31;
  int ty = threadIdx.x >> 5;       // 0..7
  int c0 = blockIdx.x * 32;
  int r0 = blockIdx.y * 32;
#pragma unroll
  for (int i = 0; i < 4; ++i) {
    int r = r0 + ty + i * 8;
    tile[ty + i * 8][tx] = f32_to_bf16(src[(size_t)r * C + c0 + tx]);
  }
  __syncthreads();
  // paired stores: 2 consecutive R-elements per u32 (coalesced)
  int tx2 = (threadIdx.x & 15) * 2;   // r-pair base 0..30
  int ty2 = threadIdx.x >> 4;         // 0..15
#pragma unroll
  for (int i = 0; i < 2; ++i) {
    int cl = ty2 + i * 16;
    unsigned v = (unsigned)tile[tx2][cl] | ((unsigned)tile[tx2 + 1][cl] << 16);
    *(unsigned*)(dst + (size_t)(c0 + cl) * R + r0 + tx2) = v;
  }
}

// ---------------------------------------------------- GEMM1: h = gelu(xW1+b1)
// A: xbf [T][D] bf16 (row gather)   B: w1T [E][F][D] bf16 (rows along K=d)
__global__ void __launch_bounds__(256)
moe_ffn1(const unsigned short* __restrict__ xbf,
         const unsigned short* __restrict__ w1T,
         const float* __restrict__ b1,   // [E][F]
         const int*   __restrict__ counts,
         const int*   __restrict__ tokenList,
         unsigned short* __restrict__ h) // [NSLOTS][F] bf16
{
  __shared__ unsigned short Alds[128 * AST];
  __shared__ unsigned short Blds[128 * AST];

  int e  = blockIdx.x >> 5;
  int mt = blockIdx.x & 31;
  int cnt = counts[e];
  int m0 = mt * 128;
  if (m0 >= cnt) return;
  int rem = cnt - m0;               // >= 1
  int nt0 = blockIdx.y * 128;

  const int* list = tokenList + e * T_TOKENS + m0;
  const unsigned short* WT = w1T + (size_t)e * D_FF * D_MODEL;

  int tid  = threadIdx.x;
  int lane = tid & 31;
  int wave = tid >> 5;
  int waveM = (wave & 3) * 32;
  int waveN = (wave >> 2) * 64;

  int arow  = tid >> 1;
  int ahalf = (tid & 1) * 16;
  int crow  = (arow < rem) ? arow : (rem - 1);   // clamp: no divergent zero-fill
  int aslot = list[crow];
  const unsigned short* asrc0 = xbf + (size_t)(aslot >> 1) * D_MODEL + ahalf;

  int bn    = tid & 127;
  int bhalf = (tid >> 7) * 16;
  const unsigned short* bsrc0 = WT + (size_t)(nt0 + bn) * D_MODEL + bhalf;

  v8f zero = {0.f,0.f,0.f,0.f,0.f,0.f,0.f,0.f};
  v8f acc[2][4];
#pragma unroll
  for (int i = 0; i < 2; ++i)
#pragma unroll
    for (int j = 0; j < 4; ++j) acc[i][j] = zero;

  for (int k0 = 0; k0 < D_MODEL; k0 += 32) {
    stage32B(asrc0 + k0, &Alds[arow * AST + ahalf]);
    stage32B(bsrc0 + k0, &Blds[bn   * AST + bhalf]);
    if (k0 + 32 < D_MODEL) {
      __builtin_prefetch(asrc0 + k0 + 32, 0, 1);
      __builtin_prefetch(bsrc0 + k0 + 32, 0, 1);
    }
    stage_wait();
    __syncthreads();

    int rsel = lane & 15;
    int akb  = (lane >> 4) ? 16 : 0;   // A: lanes16-31 start at K=8 (byte 16)
    int bkb  = (lane >> 4) ? 16 : 0;   // B: lanes16-31 start at K=8 (byte 16)
#pragma unroll
    for (int mi = 0; mi < 2; ++mi) {
      FragB16 af;
      const char* ap = (const char*)&Alds[(waveM + mi*16 + rsel) * AST];
      af.u[0] = *(const uint4*)(ap + akb);
      af.u[1] = *(const uint4*)(ap + akb + 32);
#pragma unroll
      for (int ni = 0; ni < 4; ++ni) {
        FragB16 bf;
        const char* bp = (const char*)&Blds[(waveN + ni*16 + rsel) * AST];
        bf.u[0] = *(const uint4*)(bp + bkb);
        bf.u[1] = *(const uint4*)(bp + bkb + 32);
        acc[mi][ni] = __builtin_amdgcn_wmma_f32_16x16x32_bf16(
            false, af.v, false, bf.v, (short)0, acc[mi][ni], false, false);
      }
    }
    __syncthreads();
  }

  // epilogue: + b1, exact GELU, store bf16
  int nlane = lane & 15;
  int mbase = (lane >> 4) * 8;
#pragma unroll
  for (int mi = 0; mi < 2; ++mi) {
#pragma unroll
    for (int ni = 0; ni < 4; ++ni) {
      int ncol = nt0 + waveN + ni * 16 + nlane;
      float bias = b1[e * D_FF + ncol];
#pragma unroll
      for (int r = 0; r < 8; ++r) {
        int mrow = waveM + mi * 16 + mbase + r;
        if (mrow < rem) {
          int slot = list[mrow];
          float v = acc[mi][ni][r] + bias;
          v = 0.5f * v * (1.0f + erff(v * 0.70710678118f));
          h[(size_t)slot * D_FF + ncol] = f32_to_bf16(v);
        }
      }
    }
  }
}

// ------------------------------------------- GEMM2: y = wSlot*(h W2 + b2) ---
// A: h [NSLOTS][F] bf16 (row gather)   B: w2T [E][D][F] bf16 (rows along K=f)
__global__ void __launch_bounds__(256)
moe_ffn2(const unsigned short* __restrict__ h,
         const unsigned short* __restrict__ w2T,
         const float* __restrict__ b2,           // [E][D]
         const int*   __restrict__ counts,
         const int*   __restrict__ tokenList,
         const float* __restrict__ wSlot,
         float* __restrict__ y)                  // [NSLOTS][D]
{
  __shared__ unsigned short Alds[128 * AST];
  __shared__ unsigned short Blds[128 * AST];

  int e  = blockIdx.x >> 5;
  int mt = blockIdx.x & 31;
  int cnt = counts[e];
  int m0 = mt * 128;
  if (m0 >= cnt) return;
  int rem = cnt - m0;
  int nt0 = blockIdx.y * 128;

  const int* list = tokenList + e * T_TOKENS + m0;
  const unsigned short* WT = w2T + (size_t)e * D_FF * D_MODEL;

  int tid  = threadIdx.x;
  int lane = tid & 31;
  int wave = tid >> 5;
  int waveM = (wave & 3) * 32;
  int waveN = (wave >> 2) * 64;

  int arow  = tid >> 1;
  int ahalf = (tid & 1) * 16;
  int crow  = (arow < rem) ? arow : (rem - 1);
  int aslot = list[crow];
  const unsigned short* asrc0 = h + (size_t)aslot * D_FF + ahalf;

  int bn    = tid & 127;
  int bhalf = (tid >> 7) * 16;
  const unsigned short* bsrc0 = WT + (size_t)(nt0 + bn) * D_FF + bhalf;

  v8f zero = {0.f,0.f,0.f,0.f,0.f,0.f,0.f,0.f};
  v8f acc[2][4];
#pragma unroll
  for (int i = 0; i < 2; ++i)
#pragma unroll
    for (int j = 0; j < 4; ++j) acc[i][j] = zero;

  for (int k0 = 0; k0 < D_FF; k0 += 32) {
    stage32B(asrc0 + k0, &Alds[arow * AST + ahalf]);
    stage32B(bsrc0 + k0, &Blds[bn   * AST + bhalf]);
    if (k0 + 32 < D_FF) {
      __builtin_prefetch(asrc0 + k0 + 32, 0, 1);
      __builtin_prefetch(bsrc0 + k0 + 32, 0, 1);
    }
    stage_wait();
    __syncthreads();

    int rsel = lane & 15;
    int akb  = (lane >> 4) ? 16 : 0;
    int bkb  = (lane >> 4) ? 16 : 0;
#pragma unroll
    for (int mi = 0; mi < 2; ++mi) {
      FragB16 af;
      const char* ap = (const char*)&Alds[(waveM + mi*16 + rsel) * AST];
      af.u[0] = *(const uint4*)(ap + akb);
      af.u[1] = *(const uint4*)(ap + akb + 32);
#pragma unroll
      for (int ni = 0; ni < 4; ++ni) {
        FragB16 bf;
        const char* bp = (const char*)&Blds[(waveN + ni*16 + rsel) * AST];
        bf.u[0] = *(const uint4*)(bp + bkb);
        bf.u[1] = *(const uint4*)(bp + bkb + 32);
        acc[mi][ni] = __builtin_amdgcn_wmma_f32_16x16x32_bf16(
            false, af.v, false, bf.v, (short)0, acc[mi][ni], false, false);
      }
    }
    __syncthreads();
  }

  int nlane = lane & 15;
  int mbase = (lane >> 4) * 8;
#pragma unroll
  for (int mi = 0; mi < 2; ++mi) {
#pragma unroll
    for (int ni = 0; ni < 4; ++ni) {
      int ncol = nt0 + waveN + ni * 16 + nlane;
      float bias = b2[e * D_MODEL + ncol];
#pragma unroll
      for (int r = 0; r < 8; ++r) {
        int mrow = waveM + mi * 16 + mbase + r;
        if (mrow < rem) {
          int slot = list[mrow];
          float wt = wSlot[slot];
          y[(size_t)slot * D_MODEL + ncol] = (acc[mi][ni][r] + bias) * wt;
        }
      }
    }
  }
}

// ---------------------------------------------------------------- combine ---
__global__ void __launch_bounds__(256)
combine_kernel(const float* __restrict__ y, float* __restrict__ out)
{
  size_t i = (size_t)blockIdx.x * blockDim.x + threadIdx.x; // over T*D
  size_t t = i >> 10;
  size_t d = i & 1023;
  out[i] = y[(t << 11) + d] + y[(t << 11) + 1024 + d];
}

// ---------------------------------------------------------------- launch ----
extern "C" void kernel_launch(void* const* d_in, const int* in_sizes, int n_in,
                              void* d_out, int out_size, void* d_ws, size_t ws_size,
                              hipStream_t stream) {
  const float* x      = (const float*)d_in[0];
  const float* gate_w = (const float*)d_in[1];
  const float* gate_b = (const float*)d_in[2];
  const float* w1     = (const float*)d_in[3];
  const float* b1     = (const float*)d_in[4];
  const float* w2     = (const float*)d_in[5];
  const float* b2     = (const float*)d_in[6];
  float* out = (float*)d_out;

  char* ws = (char*)d_ws;
  int*   counts    = (int*)(ws + 0);                           // 32 B
  int*   tokenList = (int*)(ws + 1024);                        // 128 KB
  float* wSlot     = (float*)(ws + 132096);                    // 32 KB
  float* partial   = (float*)(ws + 164864);                    // 16 KB
  unsigned short* xbf  = (unsigned short*)(ws + 1048576ull);   //  8 MB
  unsigned short* w1T  = (unsigned short*)(ws + 9437184ull);   // 64 MB
  unsigned short* w2T  = (unsigned short*)(ws + 76546048ull);  // 64 MB
  unsigned short* hbuf = (unsigned short*)(ws + 143654912ull); // 64 MB
  float*          ybuf = (float*)(ws + 210763776ull);          // 32 MB

  (void)hipMemsetAsync(counts, 0, NE * sizeof(int), stream);

  router_kernel<<<T_TOKENS / 8, 256, 0, stream>>>(
      x, gate_w, gate_b, counts, tokenList, wSlot, partial);

  loss_kernel<<<1, 32, 0, stream>>>(
      partial, out + (size_t)T_TOKENS * D_MODEL, T_TOKENS / 8);

  convert_x_kernel<<<(T_TOKENS * D_MODEL) / (256 * 4), 256, 0, stream>>>(x, xbf);

  // w1 [E][1024][4096] -> w1T [E][4096][1024]
  transpose_convert_kernel<<<dim3(D_FF / 32, D_MODEL / 32, NE), 256, 0, stream>>>(
      w1, w1T, D_MODEL, D_FF);
  // w2 [E][4096][1024] -> w2T [E][1024][4096]
  transpose_convert_kernel<<<dim3(D_MODEL / 32, D_FF / 32, NE), 256, 0, stream>>>(
      w2, w2T, D_FF, D_MODEL);

  moe_ffn1<<<dim3(NE * 32, D_FF / 128), 256, 0, stream>>>(
      xbf, w1T, b1, counts, tokenList, hbuf);

  moe_ffn2<<<dim3(NE * 32, D_MODEL / 128), 256, 0, stream>>>(
      hbuf, w2T, b2, counts, tokenList, wSlot, ybuf);

  combine_kernel<<<(T_TOKENS * D_MODEL) / 256, 256, 0, stream>>>(ybuf, out);
}
